// GroupedQueryAttention_29265907155282
// MI455X (gfx1250) — matrix operand order repaired and checked by reading, not compiled, not run on hardware
//
#include <hip/hip_runtime.h>
#include <hip/hip_bf16.h>

// ---------------------------------------------------------------------------
// GQA prefill for MI455X (gfx1250, wave32, WMMA, async-to-LDS staging).
// Pipeline: f32->bf16 cvt -> WMMA QKV proj (async LDS double-buffered B) ->
//           RMSNorm+RoPE -> V transpose -> flash-style causal GQA attention
//           (WMMA) -> WMMA out proj (f32).
// ---------------------------------------------------------------------------

#define NN 2048
#define CC 2048
#define HH 32
#define KVH 8
#define DD 64
#define GG (HH / KVH)   // 4

typedef __attribute__((ext_vector_type(16))) __bf16 v16bf;
typedef __attribute__((ext_vector_type(8)))  float  v8f;
typedef unsigned short u16;

union Frag16 {                 // 16 bf16 elements = 32 bytes, 16B-aligned loads
    v16bf v;
    u16   s[16];
    uint4 q[2];
};

static __device__ __forceinline__ u16 f32_to_bf16(float f) {
    unsigned int u = __float_as_uint(f);
    unsigned int r = 0x7FFFu + ((u >> 16) & 1u);   // round to nearest even
    return (u16)((u + r) >> 16);
}

static __device__ __forceinline__ v8f wmma_bf16(const Frag16& a, const Frag16& b, v8f c) {
    return __builtin_amdgcn_wmma_f32_16x16x32_bf16(false, a.v, false, b.v,
                                                   (short)0, c, false, false);
}

// async copy 16B global -> LDS (GV mode), tracked by ASYNCcnt
static __device__ __forceinline__ void async_g2l_b128(unsigned lds_addr, const void* gaddr) {
    asm volatile("global_load_async_to_lds_b128 %0, %1, off"
                 :: "v"(lds_addr), "v"((unsigned long long)(size_t)gaddr)
                 : "memory");
}

// ---------------------------------------------------------------------------
// fp32 -> bf16 elementwise conversion
// ---------------------------------------------------------------------------
__global__ void k_cvt_bf16(const float* __restrict__ in, u16* __restrict__ out, int n) {
    int i = blockIdx.x * blockDim.x + threadIdx.x;
    if (i < n) out[i] = f32_to_bf16(in[i]);
}

// ---------------------------------------------------------------------------
// NT GEMM: C[M,N] = A[M,K] * B[N,K]^T   (A,B bf16 row-major, C f32 row-major)
// block = 128 threads (4 waves) -> 64x64 C tile; wave w owns rows w*16..+15.
// B K-chunks (64 rows x 32 k = 4KB) staged through LDS with
// global_load_async_to_lds_b128, double-buffered, shared by all 4 waves.
// grid = (N/64, M/64)
// ---------------------------------------------------------------------------
__global__ void k_gemm_nt_bf16(const u16* __restrict__ A, const u16* __restrict__ B,
                               float* __restrict__ C, int M, int N, int K) {
    __shared__ __align__(16) u16 bsh[2][64 * 32];

    const int t    = threadIdx.x;
    const int lane = t & 31;
    const int wv   = t >> 5;
    const int half = lane >> 4;
    const int col  = lane & 15;
    const int m0   = blockIdx.y * 64 + wv * 16;
    const int n0   = blockIdx.x * 64;

    // this thread moves 16B chunks #t and #(t+128) of each 4KB B tile
    const int r0 = t >> 2;              // B row 0..31
    const int r1 = (t + 128) >> 2;      // B row 32..63
    const int bo = (t & 3) * 16;        // byte offset within 64B row

    const char* Bb = (const char*)B;
    const size_t rowbytes = (size_t)K * 2;

    v8f acc[4] = {};
    const u16* arow = A + (size_t)(m0 + col) * K;

    auto issue_tile = [&](int kc, int buf) {
        const char* gb = Bb + (size_t)n0 * rowbytes + (size_t)kc * 2;
        unsigned ldsb = (unsigned)(size_t)(&bsh[buf][0]);
        async_g2l_b128(ldsb + (unsigned)(r0 * 64 + bo), gb + (size_t)r0 * rowbytes + bo);
        async_g2l_b128(ldsb + (unsigned)(r1 * 64 + bo), gb + (size_t)r1 * rowbytes + bo);
    };

    const int NT = K / 32;
    issue_tile(0, 0);
    for (int i = 0; i < NT; ++i) {
        if (i + 1 < NT) {
            issue_tile((i + 1) * 32, (i + 1) & 1);
            asm volatile("s_wait_asynccnt 0x2" ::: "memory");  // tile i landed
        } else {
            asm volatile("s_wait_asynccnt 0x0" ::: "memory");
        }
        __syncthreads();                                       // visible to all waves

        Frag16 a;
        const u16* ap = arow + i * 32 + half * 16;
        a.q[0] = *(const uint4*)(ap);
        a.q[1] = *(const uint4*)(ap + 8);
        __builtin_prefetch(ap + 32, 0, 1);                     // next A chunk

        const u16* bt = &bsh[i & 1][0];
#pragma unroll
        for (int j = 0; j < 4; ++j) {
            Frag16 b;
            const u16* bp = bt + (j * 16 + col) * 32 + half * 16;
            b.q[0] = *(const uint4*)(bp);
            b.q[1] = *(const uint4*)(bp + 8);
            acc[j] = wmma_bf16(a, b, acc[j]);
        }
        __syncthreads();                                       // done reading buf before reuse
    }
#pragma unroll
    for (int j = 0; j < 4; ++j)
#pragma unroll
        for (int r = 0; r < 8; ++r)
            C[(size_t)(m0 + r + 8 * half) * N + (n0 + j * 16 + col)] = acc[j][r];
}

// ---------------------------------------------------------------------------
// Fused RMSNorm (over WIDTH) + interleaved-pair RoPE, f32 in -> bf16 out.
// one block (256 threads = 8 waves) per sequence row.
// ---------------------------------------------------------------------------
template <int WIDTH>
__global__ void k_norm_rope(const float* __restrict__ in, const float* __restrict__ w,
                            u16* __restrict__ out) {
    constexpr int E = WIDTH / 256;       // elements per thread (8 for q, 2 for k)
    const int n = blockIdx.x;
    const int t = threadIdx.x;
    const float* row = in + (size_t)n * WIDTH;

    float vals[E];
    float ss = 0.f;
#pragma unroll
    for (int i = 0; i < E; ++i) {
        vals[i] = row[t * E + i];
        ss += vals[i] * vals[i];
    }
#pragma unroll
    for (int off = 16; off > 0; off >>= 1) ss += __shfl_xor(ss, off, 32);
    __shared__ float wsum[8];
    if ((t & 31) == 0) wsum[t >> 5] = ss;
    __syncthreads();
    float tot = 0.f;
#pragma unroll
    for (int i = 0; i < 8; ++i) tot += wsum[i];
    const float inv = rsqrtf(tot / (float)WIDTH + 1e-6f);

#pragma unroll
    for (int i = 0; i < E; i += 2) {
        int   c    = t * E + i;
        int   d    = c & (DD - 1);           // position within head
        int   pair = d >> 1;
        float freq = __powf(10000.0f, -(float)(2 * pair) * (1.0f / (float)DD));
        float sn, cs;
        __sincosf((float)n * freq, &sn, &cs);
        float x1 = vals[i]     * inv * w[c];
        float x2 = vals[i + 1] * inv * w[c + 1];
        out[(size_t)n * WIDTH + c]     = f32_to_bf16(x1 * cs - x2 * sn);
        out[(size_t)n * WIDTH + c + 1] = f32_to_bf16(x1 * sn + x2 * cs);
    }
}

// ---------------------------------------------------------------------------
// V prep: vT[kv*64+d, n] = bf16(v[n, kv*64+d])   (so P*V becomes an NT gemm)
// ---------------------------------------------------------------------------
__global__ void k_vprep(const float* __restrict__ vf, u16* __restrict__ vt) {
    int idx = blockIdx.x * blockDim.x + threadIdx.x;
    if (idx >= (KVH * DD) * NN) return;
    int row = idx >> 11;       // kv*64+d
    int nn  = idx & (NN - 1);
    vt[idx] = f32_to_bf16(vf[(size_t)nn * (KVH * DD) + row]);
}

// ---------------------------------------------------------------------------
// Flash-style causal GQA attention.
// grid = (N/16 query tiles, H heads), block = 32 (one wave).
// qb: [N, H*D] bf16 (post norm+rope), kb: [N, KV*D] bf16, vt: [KV*D, N] bf16
// ob: [N, H*D] bf16 attention output.
// ---------------------------------------------------------------------------
__global__ void k_attn(const u16* __restrict__ qb, const u16* __restrict__ kb,
                       const u16* __restrict__ vt, u16* __restrict__ ob) {
    const int lane = threadIdx.x & 31;
    const int half = lane >> 4;
    const int col  = lane & 15;
    const int nq0  = blockIdx.x * 16;
    const int h    = blockIdx.y;
    const int kvh  = h >> 2;                      // G = 4

    __shared__ __align__(16) u16 pls[16 * 32];    // P tile staging (16n x 32m)

    v8f   o[4] = {};
    float mi[8], li[8];
#pragma unroll
    for (int r = 0; r < 8; ++r) { mi[r] = -1e30f; li[r] = 0.f; }

    const u16* qrow = qb + (size_t)(nq0 + col) * (HH * DD) + h * DD;

    for (int m0 = 0; m0 < nq0 + 16; m0 += 32) {
        // ---- scores S[16n x 32m] = Q * K^T over D=64 ----
        v8f s[2] = {};
#pragma unroll
        for (int kc = 0; kc < DD; kc += 32) {
            Frag16 a;
            const u16* ap = qrow + kc + half * 16;
            a.q[0] = *(const uint4*)(ap);
            a.q[1] = *(const uint4*)(ap + 8);
#pragma unroll
            for (int j = 0; j < 2; ++j) {
                Frag16 b;
                const u16* bp = kb + (size_t)(m0 + j * 16 + col) * (KVH * DD)
                                + kvh * DD + kc + half * 16;
                b.q[0] = *(const uint4*)(bp);
                b.q[1] = *(const uint4*)(bp + 8);
                s[j] = wmma_bf16(a, b, s[j]);
            }
        }
        // ---- scale + causal mask + online softmax (row stats per lane-half) ----
#pragma unroll
        for (int r = 0; r < 8; ++r) {
            const int nrow = nq0 + r + 8 * half;
            float v0 = s[0][r] * 0.125f;
            float v1 = s[1][r] * 0.125f;
            if (m0 + col      > nrow) v0 = -1e30f;
            if (m0 + 16 + col > nrow) v1 = -1e30f;
            float mx = fmaxf(v0, v1);
#pragma unroll
            for (int off = 1; off < 16; off <<= 1) mx = fmaxf(mx, __shfl_xor(mx, off, 32));
            float mnew  = fmaxf(mi[r], mx);
            float alpha = __expf(mi[r] - mnew);
            float p0    = __expf(v0 - mnew);
            float p1    = __expf(v1 - mnew);
            float rs    = p0 + p1;
#pragma unroll
            for (int off = 1; off < 16; off <<= 1) rs += __shfl_xor(rs, off, 32);
            li[r] = li[r] * alpha + rs;
            mi[r] = mnew;
#pragma unroll
            for (int dt = 0; dt < 4; ++dt) o[dt][r] *= alpha;
            const int nl = r + 8 * half;
            pls[nl * 32 + col]      = f32_to_bf16(p0);
            pls[nl * 32 + 16 + col] = f32_to_bf16(p1);
        }
        asm volatile("s_wait_dscnt 0x0" ::: "memory");   // LDS store->load, same wave

        // ---- O += P * V   (Vt rows are the d dimension -> NT gemm) ----
        Frag16 pa;
        {
            const u16* pp = &pls[col * 32 + half * 16];
            pa.q[0] = *(const uint4*)(pp);
            pa.q[1] = *(const uint4*)(pp + 8);
        }
#pragma unroll
        for (int dt = 0; dt < 4; ++dt) {
            Frag16 b;
            const u16* vp = vt + (size_t)(kvh * DD + dt * 16 + col) * NN + m0 + half * 16;
            b.q[0] = *(const uint4*)(vp);
            b.q[1] = *(const uint4*)(vp + 8);
            o[dt] = wmma_bf16(pa, b, o[dt]);
        }
    }
    // ---- epilogue: normalize, write bf16 ----
#pragma unroll
    for (int r = 0; r < 8; ++r) {
        const float invl = 1.0f / li[r];
        const int   nrow = nq0 + r + 8 * half;
#pragma unroll
        for (int dt = 0; dt < 4; ++dt)
            ob[(size_t)nrow * (HH * DD) + h * DD + dt * 16 + col] =
                f32_to_bf16(o[dt][r] * invl);
    }
}

// ---------------------------------------------------------------------------
extern "C" void kernel_launch(void* const* d_in, const int* in_sizes, int n_in,
                              void* d_out, int out_size, void* d_ws, size_t ws_size,
                              hipStream_t stream) {
    const float* x  = (const float*)d_in[0];
    const float* wq = (const float*)d_in[1];
    const float* wk = (const float*)d_in[2];
    const float* wv = (const float*)d_in[3];
    const float* wo = (const float*)d_in[4];
    const float* qw = (const float*)d_in[5];
    const float* kw = (const float*)d_in[6];
    float* out = (float*)d_out;

    char* p = (char*)d_ws;
    auto alloc = [&](size_t bytes) {
        char* r = p;
        p += (bytes + 255) & ~(size_t)255;
        return r;
    };
    u16*   xb  = (u16*)  alloc((size_t)NN * CC * 2);          // x bf16
    u16*   wqb = (u16*)  alloc((size_t)CC * CC * 2);          // wq bf16
    u16*   wkb = (u16*)  alloc((size_t)KVH * DD * CC * 2);    // wk bf16
    u16*   wvb = (u16*)  alloc((size_t)KVH * DD * CC * 2);    // wv bf16
    u16*   wob = (u16*)  alloc((size_t)CC * CC * 2);          // wo bf16
    float* qf  = (float*)alloc((size_t)NN * HH * DD * 4);     // q proj f32
    float* kf  = (float*)alloc((size_t)NN * KVH * DD * 4);    // k proj f32
    float* vf  = (float*)alloc((size_t)NN * KVH * DD * 4);    // v proj f32
    u16*   qb  = (u16*)  alloc((size_t)NN * HH * DD * 2);     // q bf16 (norm+rope)
    u16*   kb  = (u16*)  alloc((size_t)NN * KVH * DD * 2);    // k bf16 (norm+rope)
    u16*   vt  = (u16*)  alloc((size_t)KVH * DD * NN * 2);    // v^T bf16
    u16*   ob  = (u16*)  alloc((size_t)NN * HH * DD * 2);     // attn out bf16

    const int T = 256;
    auto cvt = [&](const float* src, u16* dst, int n) {
        k_cvt_bf16<<<(n + T - 1) / T, T, 0, stream>>>(src, dst, n);
    };
    cvt(x,  xb,  NN * CC);
    cvt(wq, wqb, CC * CC);
    cvt(wk, wkb, KVH * DD * CC);
    cvt(wv, wvb, KVH * DD * CC);
    cvt(wo, wob, CC * CC);

    // QKV projections (NT gemms, async-LDS staged B)
    k_gemm_nt_bf16<<<dim3(CC / 64, NN / 64), 128, 0, stream>>>(xb, wqb, qf, NN, CC, CC);
    k_gemm_nt_bf16<<<dim3((KVH * DD) / 64, NN / 64), 128, 0, stream>>>(xb, wkb, kf, NN, KVH * DD, CC);
    k_gemm_nt_bf16<<<dim3((KVH * DD) / 64, NN / 64), 128, 0, stream>>>(xb, wvb, vf, NN, KVH * DD, CC);

    // RMSNorm + RoPE
    k_norm_rope<HH * DD> <<<NN, 256, 0, stream>>>(qf, qw, qb);
    k_norm_rope<KVH * DD><<<NN, 256, 0, stream>>>(kf, kw, kb);

    // V transpose to bf16
    k_vprep<<<(KVH * DD * NN + T - 1) / T, T, 0, stream>>>(vf, vt);

    // causal GQA attention
    k_attn<<<dim3(NN / 16, HH), 32, 0, stream>>>(qb, kb, vt, ob);

    // output projection straight into f32 d_out
    k_gemm_nt_bf16<<<dim3(CC / 64, NN / 64), 128, 0, stream>>>(ob, wob, out, NN, CC, CC);
}
